// PPIHetero_26482768347972
// MI455X (gfx1250) — compile-verified, matrix operand order checked by reading, not run on hardware
//
#include <hip/hip_runtime.h>
#include <hip/hip_bf16.h>

typedef __attribute__((ext_vector_type(16))) __bf16 v16bf;
typedef __attribute__((ext_vector_type(8)))  __bf16 v8bf;
typedef __attribute__((ext_vector_type(2)))  __bf16 v2bf;
typedef __attribute__((ext_vector_type(8)))  float  v8f;

#define WPB 8   // waves per block (256 threads, wave32)
#define NT  8   // N-tiles (16x16) per wave; A conversion amortized NT x

__device__ __forceinline__ void sched_fence() {
#if __has_builtin(__builtin_amdgcn_sched_barrier)
    __builtin_amdgcn_sched_barrier(0);
#endif
}

__device__ __forceinline__ __bf16 f2bf(float f) {
    unsigned u = __float_as_uint(f);
    u += 0x7FFFu + ((u >> 16) & 1u);            // round to nearest even
    unsigned short h = (unsigned short)(u >> 16);
    return __builtin_bit_cast(__bf16, h);
}

// packed f32x2 -> bf16x2 (hardware cvt when toolchain exposes it)
__device__ __forceinline__ v2bf pk2(float lo, float hi) {
#if __has_builtin(__builtin_amdgcn_cvt_pk_bf16_f32)
    auto r = __builtin_amdgcn_cvt_pk_bf16_f32(lo, hi);
    return __builtin_bit_cast(v2bf, r);
#else
    v2bf r; r[0] = f2bf(lo); r[1] = f2bf(hi); return r;
#endif
}

// ---------------- fill ----------------
__global__ void k_fill0(float* __restrict__ p, long n) {
    long i = (long)blockIdx.x * blockDim.x + threadIdx.x;
    long stride = (long)gridDim.x * blockDim.x;
    for (; i < n; i += stride) p[i] = 0.0f;
}

// ---------------- f32 -> bf16 ----------------
__global__ void k_cvt_bf16(const float* __restrict__ in, __bf16* __restrict__ out, long n) {
    long i = (long)blockIdx.x * blockDim.x + threadIdx.x;
    long stride = (long)gridDim.x * blockDim.x;
    for (; i < n; i += stride) out[i] = f2bf(in[i]);
}

// ---------------- segment scatter-add (one block per edge, 64 thr x float4) ----------------
__global__ void k_scatter_add(const float* __restrict__ src, const int* __restrict__ ei,
                              float* __restrict__ agg, float* __restrict__ cnt, int E) {
    int e = blockIdx.x;
    int s = ei[e];
    int d = ei[E + e];
    const float4* sp = (const float4*)(src + (size_t)s * 256);
    float* dp = agg + (size_t)d * 256;
    int t = threadIdx.x;                 // 0..63
    float4 v = sp[t];
    atomicAdd(dp + t * 4 + 0, v.x);
    atomicAdd(dp + t * 4 + 1, v.y);
    atomicAdd(dp + t * 4 + 2, v.z);
    atomicAdd(dp + t * 4 + 3, v.w);
    if (t == 0) atomicAdd(cnt + d, 1.0f);
}

// ---------------- divide rows by max(count,1) ----------------
__global__ void k_seg_norm(float* __restrict__ agg, const float* __restrict__ cnt, long total) {
    long i = (long)blockIdx.x * blockDim.x + threadIdx.x;
    if (i >= total) return;
    float c = cnt[i >> 8];               // H == 256
    agg[i] = agg[i] / fmaxf(c, 1.0f);
}

// ---------------- WMMA GEMM: C[M,N] (+)= A[M,K] @ Bw[N,K]^T (+bias)(+relu) ----------------
// Compile-time N,K. Per K-step: A loads, 16 immediate-offset b128 B loads in flight,
// A convert overlaps B latency, then two half-batches of 4 back-to-back v_wmma with
// staged s_wait_loadcnt. __launch_bounds__(...,1) raises the VGPR budget so the
// ~160-VGPR live set (64 acc + B fragments + staging) does not spill.
// flags: bit0 = accumulate into existing C, bit1 = relu at end
template<int N, int K>
__global__ void __launch_bounds__(WPB * 32, 1)
k_gemm_bf16w(const float* __restrict__ A, const __bf16* __restrict__ Bw,
             const float* __restrict__ bias, float* __restrict__ C,
             int M, int flags) {
    const int wave = threadIdx.x >> 5;
    const int lane = threadIdx.x & 31;
    constexpr int ngroups = (N >> 4) / NT;          // groups of NT n-tiles
    const int tile = blockIdx.x * WPB + wave;
    if (tile >= (M >> 4) * ngroups) return;         // uniform per wave
    const int mt = tile / ngroups;
    const int nt0 = (tile - mt * ngroups) * NT;
    const int half = lane >> 4;                     // 0: k 0-7,16-23; 1: k 8-15,24-31
    const int l = lane & 15;
    const int kb = half * 8;
    const float*  aBase = A  + (size_t)(mt * 16 + l) * K + kb;
    const __bf16* bBase = Bw + (size_t)(nt0 * 16 + l) * K + kb;
    const int c_row0 = mt * 16 + half * 8;
    float* cBase = C + (size_t)c_row0 * N;

    v8f acc[NT];
    if (flags & 1) {
        #pragma unroll
        for (int t = 0; t < NT; t++) {
            const int c_col = (nt0 + t) * 16 + l;
            #pragma unroll
            for (int v = 0; v < 8; v++) acc[t][v] = cBase[v * N + c_col];
        }
    } else {
        #pragma unroll
        for (int t = 0; t < NT; t++)
            #pragma unroll
            for (int v = 0; v < 8; v++) acc[t][v] = 0.0f;
    }

    struct B2 { v8bf lo; v8bf hi; };
    for (int k0 = 0; k0 < K; k0 += 32) {
        // ---- A loads first ----
        const float* ap = aBase + k0;
        float4 f0 = *(const float4*)(ap);
        float4 f1 = *(const float4*)(ap + 4);
        float4 f2 = *(const float4*)(ap + 16);
        float4 f3 = *(const float4*)(ap + 20);
        // ---- 16 B-fragment loads, immediate offsets, batched in flight ----
        B2 bb[NT];
        #pragma unroll
        for (int t = 0; t < NT; t++) {
            bb[t].lo = *(const v8bf*)(bBase + k0 + t * (16 * K));
            bb[t].hi = *(const v8bf*)(bBase + k0 + t * (16 * K) + 16);
        }
        // ---- A conversion overlaps B load latency ----
        v2bf tmp[8];
        tmp[0] = pk2(f0.x, f0.y); tmp[1] = pk2(f0.z, f0.w);
        tmp[2] = pk2(f1.x, f1.y); tmp[3] = pk2(f1.z, f1.w);
        tmp[4] = pk2(f2.x, f2.y); tmp[5] = pk2(f2.z, f2.w);
        tmp[6] = pk2(f3.x, f3.y); tmp[7] = pk2(f3.z, f3.w);
        v16bf a = __builtin_bit_cast(v16bf, tmp);

        sched_fence();   // loads/convert stay above
        #pragma unroll
        for (int t = 0; t < NT / 2; t++) {
            v16bf b = __builtin_bit_cast(v16bf, bb[t]);
            acc[t] = __builtin_amdgcn_wmma_f32_16x16x32_bf16(
                false, a, false, b, (short)0, acc[t], false, false);
        }
        sched_fence();   // second half-batch drains remaining in-flight loads
        #pragma unroll
        for (int t = NT / 2; t < NT; t++) {
            v16bf b = __builtin_bit_cast(v16bf, bb[t]);
            acc[t] = __builtin_amdgcn_wmma_f32_16x16x32_bf16(
                false, a, false, b, (short)0, acc[t], false, false);
        }
    }

    #pragma unroll
    for (int t = 0; t < NT; t++) {
        const int c_col = (nt0 + t) * 16 + l;
        if (bias) {
            float bv = bias[c_col];
            #pragma unroll
            for (int v = 0; v < 8; v++) acc[t][v] += bv;
        }
        if (flags & 2) {
            #pragma unroll
            for (int v = 0; v < 8; v++) acc[t][v] = fmaxf(acc[t][v], 0.0f);
        }
        #pragma unroll
        for (int v = 0; v < 8; v++) cBase[v * N + c_col] = acc[t][v];
    }
}

// ---------------- edge dot-product: out[e] = dot(X[i[e]], Y[j[e]]), 1 wave / edge ----------------
__global__ void k_edge_dot(const float* __restrict__ X, const float* __restrict__ Y,
                           const int* __restrict__ eli, float* __restrict__ out, int EL) {
    int wave = threadIdx.x >> 5;
    int lane = threadIdx.x & 31;
    int e = blockIdx.x * WPB + wave;
    if (e >= EL) return;                              // uniform per wave
    int i = eli[e];
    int j = eli[EL + e];
    const float4* xp = (const float4*)(X + (size_t)i * 256) + lane * 2;
    const float4* yp = (const float4*)(Y + (size_t)j * 256) + lane * 2;
    float4 x0 = xp[0], x1 = xp[1], y0 = yp[0], y1 = yp[1];
    float s = x0.x * y0.x + x0.y * y0.y + x0.z * y0.z + x0.w * y0.w
            + x1.x * y1.x + x1.y * y1.y + x1.z * y1.z + x1.w * y1.w;
    #pragma unroll
    for (int off = 16; off > 0; off >>= 1) s += __shfl_xor(s, off, 32);
    if (lane == 0) out[e] = s;
}

extern "C" void kernel_launch(void* const* d_in, const int* in_sizes, int n_in,
                              void* d_out, int out_size, void* d_ws, size_t ws_size,
                              hipStream_t stream) {
    const int H = 256;
    const int F = in_sizes[2] / H;          // 1280
    const int NP = in_sizes[0] / F;         // 50000 peptides
    const int NR = in_sizes[1] / F;         // 50000 proteins
    const int E_pp = in_sizes[12] / 2;
    const int E_rp = in_sizes[13] / 2;
    const int E_pr = in_sizes[14] / 2;
    const int ELh  = in_sizes[15] / 2;
    const int ELo  = in_sizes[16] / 2;

    const float* x_pep  = (const float*)d_in[0];
    const float* x_prot = (const float*)d_in[1];
    const float* Wpep   = (const float*)d_in[2];
    const float* bpep   = (const float*)d_in[3];
    const float* Wprot  = (const float*)d_in[4];
    const float* bprot  = (const float*)d_in[5];
    const float* Wl1    = (const float*)d_in[6];
    const float* bl1    = (const float*)d_in[7];
    const float* Wr1    = (const float*)d_in[8];
    const float* Wl2    = (const float*)d_in[9];
    const float* bl2    = (const float*)d_in[10];
    const float* Wr2    = (const float*)d_in[11];
    const int* ei_pp    = (const int*)d_in[12];
    const int* ei_rp    = (const int*)d_in[13];
    const int* ei_prot  = (const int*)d_in[14];
    const int* eli_het  = (const int*)d_in[15];
    const int* eli_homo = (const int*)d_in[16];
    float* out = (float*)d_out;

    // ---- workspace carve ----
    char* ws = (char*)d_ws;
    const size_t NBp = (size_t)NP * H;      // 12.8M floats
    const size_t NBr = (size_t)NR * H;
    float* h_pep  = (float*)ws; ws += NBp * 4;
    float* h_prot = (float*)ws; ws += NBr * 4;
    float* p1     = (float*)ws; ws += NBp * 4;
    float* q1     = (float*)ws; ws += NBr * 4;
    float* agg    = (float*)ws; ws += ((NBp > NBr) ? NBp : NBr) * 4;
    float* cnt    = (float*)ws; ws += ((NP > NR) ? NP : NR) * 4;
    __bf16* Wpep_bf  = (__bf16*)ws; ws += (size_t)H * F * 2;
    __bf16* Wprot_bf = (__bf16*)ws; ws += (size_t)H * F * 2;
    __bf16* Wl1_bf   = (__bf16*)ws; ws += (size_t)3 * H * H * 2;
    __bf16* Wr1_bf   = (__bf16*)ws; ws += (size_t)3 * H * H * 2;
    __bf16* Wl2_bf   = (__bf16*)ws; ws += (size_t)3 * H * H * 2;
    __bf16* Wr2_bf   = (__bf16*)ws; ws += (size_t)3 * H * H * 2;
    float* p2 = h_pep;   // reuse after layer 1 consumed
    float* q2 = h_prot;

    auto cvt = [&](const float* in, __bf16* o, long n) {
        int blocks = (int)((n + 255) / 256); if (blocks > 4096) blocks = 4096;
        k_cvt_bf16<<<blocks, 256, 0, stream>>>(in, o, n);
    };
    auto gemm = [&](const float* A, const __bf16* B, const float* bias, float* C,
                    int M, int K, int flags) {
        int tiles = (M / 16) * ((256 / 16) / NT);
        int blocks = (tiles + WPB - 1) / WPB;
        if (K == 256)
            k_gemm_bf16w<256, 256><<<blocks, 32 * WPB, 0, stream>>>(A, B, bias, C, M, flags);
        else
            k_gemm_bf16w<256, 1280><<<blocks, 32 * WPB, 0, stream>>>(A, B, bias, C, M, flags);
    };
    auto segmean = [&](const float* src, const int* ei, int E_, int ndst) {
        long tot = (long)ndst * H;
        k_fill0<<<4096, 256, 0, stream>>>(agg, tot);
        k_fill0<<<256, 256, 0, stream>>>(cnt, ndst);
        k_scatter_add<<<E_, 64, 0, stream>>>(src, ei, agg, cnt, E_);
        k_seg_norm<<<(int)((tot + 255) / 256), 256, 0, stream>>>(agg, cnt, tot);
    };

    // ---- weights -> bf16 ----
    cvt(Wpep,  Wpep_bf,  (long)H * F);
    cvt(Wprot, Wprot_bf, (long)H * F);
    cvt(Wl1, Wl1_bf, (long)3 * H * H);
    cvt(Wr1, Wr1_bf, (long)3 * H * H);
    cvt(Wl2, Wl2_bf, (long)3 * H * H);
    cvt(Wr2, Wr2_bf, (long)3 * H * H);

    // ---- input projections ----
    gemm(x_pep,  Wpep_bf,  bpep,  h_pep,  NP, F, 0);
    gemm(x_prot, Wprot_bf, bprot, h_prot, NR, F, 0);

    // ---- layer 1: p1 = relu(sage(h_prot -> pep)) ----
    segmean(h_prot, ei_rp, E_rp, NP);
    gemm(agg,   Wl1_bf + (size_t)1 * H * H, bl1 + 1 * H, p1, NP, H, 0);
    gemm(h_pep, Wr1_bf + (size_t)1 * H * H, nullptr,     p1, NP, H, 1 | 2);

    // ---- layer 1: q1 = relu(sage(h_pep->prot) + sage(h_prot->prot)) ----
    segmean(h_pep, ei_pp, E_pp, NR);
    gemm(agg,    Wl1_bf + (size_t)0 * H * H, bl1 + 0 * H, q1, NR, H, 0);
    gemm(h_prot, Wr1_bf + (size_t)0 * H * H, nullptr,     q1, NR, H, 1);
    segmean(h_prot, ei_prot, E_pr, NR);
    gemm(agg,    Wl1_bf + (size_t)2 * H * H, bl1 + 2 * H, q1, NR, H, 1);
    gemm(h_prot, Wr1_bf + (size_t)2 * H * H, nullptr,     q1, NR, H, 1 | 2);

    // ---- layer 2: p2 = sage(q1 -> pep via ei_rp) ----  (p2 aliases h_pep: safe, h_pep consumed)
    segmean(q1, ei_rp, E_rp, NP);
    gemm(agg, Wl2_bf + (size_t)1 * H * H, bl2 + 1 * H, p2, NP, H, 0);
    gemm(p1,  Wr2_bf + (size_t)1 * H * H, nullptr,     p2, NP, H, 1);

    // ---- layer 2: q2 = sage(p1->prot) + sage(q1->prot) ---- (q2 aliases h_prot: safe)
    segmean(p1, ei_pp, E_pp, NR);
    gemm(agg, Wl2_bf + (size_t)0 * H * H, bl2 + 0 * H, q2, NR, H, 0);
    gemm(q1,  Wr2_bf + (size_t)0 * H * H, nullptr,     q2, NR, H, 1);
    segmean(q1, ei_prot, E_pr, NR);
    gemm(agg, Wl2_bf + (size_t)2 * H * H, bl2 + 2 * H, q2, NR, H, 1);
    gemm(q1,  Wr2_bf + (size_t)2 * H * H, nullptr,     q2, NR, H, 1);

    // ---- edge classifiers ----
    k_edge_dot<<<(ELh + WPB - 1) / WPB, 32 * WPB, 0, stream>>>(p2, q2, eli_het,  out,       ELh);
    k_edge_dot<<<(ELo + WPB - 1) / WPB, 32 * WPB, 0, stream>>>(q2, q2, eli_homo, out + ELh, ELo);
}
